// ContextPrioritizer_21217138442629
// MI455X (gfx1250) — compile-verified, matrix operand order checked by reading, not run on hardware
//
#include <hip/hip_runtime.h>
#include <math.h>

// ---------------------------------------------------------------------------
// MI455X (gfx1250) implementation: bf16 WMMA pipeline
//   stage 0: transpose+convert weights fp32 -> bf16 (N x K layout) in ws
//   stage 1: fold query half of W1 into a per-output bias (qbias)
//   stage 2: fused scorer (GEMM1 + relu + GEMM2 + relu + dot + sigmoid)
//   stage 3: histogram top-k threshold + deterministic compaction
//   stage 4: compressor: 3 WMMA GEMMs (gather -> Y1 -> Y2 -> out)
//
// Round-1 fix: batch all B-tile loads of a K-step into one clause before the
// WMMA batch (staggered s_wait_loadcnt instead of full drains), and double-
// buffer the A operand across K-steps.
// ---------------------------------------------------------------------------

typedef __attribute__((ext_vector_type(16))) __bf16 v16bf;
typedef __attribute__((ext_vector_type(8)))  __bf16 v8bf;
typedef __attribute__((ext_vector_type(8)))  float  v8f;
typedef __attribute__((ext_vector_type(4)))  float  v4f;

#define NCTX   65536
#define HDIM   2048
#define KSEL   4096

__device__ __forceinline__ v16bf bf_combine(v8bf lo, v8bf hi) {
  v16bf r;
#pragma unroll
  for (int i = 0; i < 8; ++i) { r[i] = lo[i]; r[8 + i] = hi[i]; }
  return r;
}

// 8 contiguous fp32 -> v8bf (two 16B loads + pack-convert)
__device__ __forceinline__ v8bf cvt8(const float* __restrict__ p) {
  v4f f0 = *(const v4f*)p;
  v4f f1 = *(const v4f*)(p + 4);
  v8bf r;
#pragma unroll
  for (int i = 0; i < 4; ++i) { r[i] = (__bf16)f0[i]; r[4 + i] = (__bf16)f1[i]; }
  return r;
}

__device__ __forceinline__ v16bf load16bf(const __bf16* __restrict__ p) {
  // p at K-offset multiple of 8 (16B aligned); halves: K..K+7 and K+16..K+23
  return bf_combine(*(const v8bf*)p, *(const v8bf*)(p + 16));
}

__device__ __forceinline__ int score_bin(float s) {
  int b = (int)(s * 4096.0f);
  return b < 0 ? 0 : (b > 4095 ? 4095 : b);
}

// ---------------------------------------------------------------------------
// Weight transpose + fp32->bf16:  in (K x N, row major) -> out (N x K, row major)
// ---------------------------------------------------------------------------
__global__ void transpose_bf16_kernel(const float* __restrict__ in,
                                      __bf16* __restrict__ out, int K, int N) {
  int idx = blockIdx.x * blockDim.x + threadIdx.x;
  int total = K * N;
  if (idx >= total) return;
  int k = idx / N;
  int n = idx - k * N;
  out[(size_t)n * K + k] = (__bf16)in[idx];
}

// qbias[n] = b1[n] + sum_k q[k] * W1[2048+k][n]   (query half of the concat)
__global__ void qbias_kernel(const float* __restrict__ q, const float* __restrict__ W1,
                             const float* __restrict__ b1, float* __restrict__ qb) {
  int n = blockIdx.x * blockDim.x + threadIdx.x;
  if (n >= 512) return;
  float s = b1[n];
  for (int k = 0; k < HDIM; ++k)
    s += q[k] * W1[(size_t)(HDIM + k) * 512 + n];
  qb[n] = s;
}

// ---------------------------------------------------------------------------
// Fused scorer: 16 context rows per block (4 waves).
//   h1(16x512) = relu(ctx_tile @ W1ctx + qbias)     K=2048, bf16 WMMA
//   h2(16x256) = relu(h1 @ W2 + b2)                 K=512,  bf16 WMMA (A from LDS)
//   score      = sigmoid(h2 @ W3 + b3)
// ---------------------------------------------------------------------------
__global__ __launch_bounds__(128)
void score_kernel(const float* __restrict__ ctx, const __bf16* __restrict__ W1T,
                  const float* __restrict__ qbias, const __bf16* __restrict__ W2T,
                  const float* __restrict__ b2, const float* __restrict__ W3,
                  const float* __restrict__ b3, float* __restrict__ S) {
  const int lane = threadIdx.x & 31;
  const int wid  = threadIdx.x >> 5;
  const int lm   = lane & 15;
  const int hi   = lane >> 4;
  const int row0 = blockIdx.x * 16;

  __shared__ __bf16 h1[16 * 512];
  __shared__ __bf16 h2[16 * 256];

  // ---- GEMM1: each wave owns a 16x128 strip of h1 (8 accumulators) ----
  const int cb = wid * 128;
  v8f acc[8];
#pragma unroll
  for (int j = 0; j < 8; ++j) {
    float b = qbias[cb + j * 16 + lm];
#pragma unroll
    for (int r = 0; r < 8; ++r) acc[j][r] = b;
  }
  const float*  arow  = ctx + (size_t)(row0 + lm) * HDIM;
  const __bf16* brow  = W1T + (size_t)(cb + lm) * HDIM + hi * 8;

  v16bf A = bf_combine(cvt8(arow + hi * 8), cvt8(arow + 16 + hi * 8));
  for (int kk = 0; kk < HDIM; kk += 32) {
    // one clause of 16 b128 loads: all 8 B tiles of this K-step
    v16bf Bt[8];
#pragma unroll
    for (int j = 0; j < 8; ++j)
      Bt[j] = load16bf(brow + (size_t)j * 16 * HDIM + kk);
    // next-step A loads + cvt overlap the WMMA batch below
    v16bf Anext;
    if (kk + 32 < HDIM) {
      __builtin_prefetch((const void*)(arow + kk + 64 + hi * 8), 0, 1);
      Anext = bf_combine(cvt8(arow + kk + 32 + hi * 8),
                         cvt8(arow + kk + 48 + hi * 8));
    }
#pragma unroll
    for (int j = 0; j < 8; ++j)
      acc[j] = __builtin_amdgcn_wmma_f32_16x16x32_bf16(false, A, false, Bt[j],
                                                       (short)0, acc[j], false, false);
    A = Anext;
  }
#pragma unroll
  for (int j = 0; j < 8; ++j) {
    int col = cb + j * 16 + lm;
#pragma unroll
    for (int r = 0; r < 8; ++r) {
      float v = acc[j][r];
      h1[(hi * 8 + r) * 512 + col] = (__bf16)(v > 0.0f ? v : 0.0f);
    }
  }
  __syncthreads();

  // ---- GEMM2: each wave owns a 16x64 strip of h2 (4 accumulators), K=512 ----
  const int cb2 = wid * 64;
  v8f acc2[4];
#pragma unroll
  for (int j = 0; j < 4; ++j) {
    float b = b2[cb2 + j * 16 + lm];
#pragma unroll
    for (int r = 0; r < 8; ++r) acc2[j][r] = b;
  }
  const __bf16* brow2 = W2T + (size_t)(cb2 + lm) * 512 + hi * 8;
  for (int kk = 0; kk < 512; kk += 32) {
    v16bf Bt[4];
#pragma unroll
    for (int j = 0; j < 4; ++j)
      Bt[j] = load16bf(brow2 + (size_t)j * 16 * 512 + kk);
    const __bf16* ap = &h1[lm * 512 + kk + hi * 8];
    v16bf A2 = bf_combine(*(const v8bf*)ap, *(const v8bf*)(ap + 16));
#pragma unroll
    for (int j = 0; j < 4; ++j)
      acc2[j] = __builtin_amdgcn_wmma_f32_16x16x32_bf16(false, A2, false, Bt[j],
                                                        (short)0, acc2[j], false, false);
  }
#pragma unroll
  for (int j = 0; j < 4; ++j) {
    int col = cb2 + j * 16 + lm;
#pragma unroll
    for (int r = 0; r < 8; ++r) {
      float v = acc2[j][r];
      h2[(hi * 8 + r) * 256 + col] = (__bf16)(v > 0.0f ? v : 0.0f);
    }
  }
  __syncthreads();

  // ---- scores: lane L (wave 0) handles row L ----
  if (wid == 0 && lane < 16) {
    float s = b3[0];
    const __bf16* hp = &h2[lane * 256];
    for (int k = 0; k < 256; ++k) s += (float)hp[k] * W3[k];
    S[row0 + lane] = 1.0f / (1.0f + __expf(-s));
  }
}

// ---------------------------------------------------------------------------
// Top-k = 4096 via histogram threshold + deterministic compaction
// ---------------------------------------------------------------------------
__global__ void zero_kernel(int* __restrict__ p, int n) {
  int i = blockIdx.x * blockDim.x + threadIdx.x;
  if (i < n) p[i] = 0;
}

__global__ void hist_kernel(const float* __restrict__ S, int* __restrict__ bins) {
  int i = blockIdx.x * blockDim.x + threadIdx.x;
  atomicAdd(&bins[score_bin(S[i])], 1);
}

__global__ void thresh_kernel(const int* __restrict__ bins, int* __restrict__ meta) {
  if (threadIdx.x != 0 || blockIdx.x != 0) return;
  int cnt = 0, b = 4095;
  for (; b >= 0; --b) {
    if (cnt + bins[b] >= KSEL) break;
    cnt += bins[b];
  }
  if (b < 0) b = 0;
  meta[0] = b;            // threshold bin
  meta[1] = cnt;          // strictly above threshold bin
  meta[2] = KSEL - cnt;   // to take from threshold bin
}

__global__ __launch_bounds__(32)
void count_kernel(const float* __restrict__ S, const int* __restrict__ meta,
                  int* __restrict__ cntA, int* __restrict__ cntB) {
  int lane = threadIdx.x;
  int base = blockIdx.x * 1024;
  int binT = meta[0];
  int a = 0, c = 0;
  for (int i = lane; i < 1024; i += 32) {
    int b = score_bin(S[base + i]);
    if (b > binT) a++;
    else if (b == binT) c++;
  }
#pragma unroll
  for (int off = 16; off > 0; off >>= 1) {
    a += __shfl_down(a, off, 32);
    c += __shfl_down(c, off, 32);
  }
  if (lane == 0) { cntA[blockIdx.x] = a; cntB[blockIdx.x] = c; }
}

__global__ void scan_kernel(const int* __restrict__ cntA, const int* __restrict__ cntB,
                            int* __restrict__ offA, int* __restrict__ offB) {
  if (threadIdx.x != 0 || blockIdx.x != 0) return;
  int sa = 0, sb = 0;
  for (int i = 0; i < 64; ++i) {
    offA[i] = sa; sa += cntA[i];
    offB[i] = sb; sb += cntB[i];
  }
}

__global__ __launch_bounds__(32)
void select_kernel(const float* __restrict__ S, const int* __restrict__ meta,
                   const int* __restrict__ offA, const int* __restrict__ offB,
                   int* __restrict__ sel) {
  int lane = threadIdx.x;
  int base = blockIdx.x * 1024;
  int binT = meta[0], totalA = meta[1];
  int oa = offA[blockIdx.x], ob = offB[blockIdx.x];
  unsigned ltmask = (1u << lane) - 1u;
  for (int i = 0; i < 1024; i += 32) {
    int idx = base + i + lane;
    int b = score_bin(S[idx]);
    unsigned mA = __builtin_amdgcn_ballot_w32(b > binT);
    unsigned mB = __builtin_amdgcn_ballot_w32(b == binT);
    if (b > binT) {
      sel[oa + __builtin_popcount(mA & ltmask)] = idx;
    } else if (b == binT) {
      int pos = totalA + ob + __builtin_popcount(mB & ltmask);
      if (pos < KSEL) sel[pos] = idx;
    }
    oa += __builtin_popcount(mA);
    ob += __builtin_popcount(mB);
  }
}

// ---------------------------------------------------------------------------
// Generic compressor GEMM: wave computes 16x64 strip; K step 32 bf16 WMMA.
//   GATHER_F32: A rows are fp32 gathered through sel[] (converted on the fly)
//   RELU_BF16_OUT: relu + bf16 output (intermediate), else raw fp32 output
// ---------------------------------------------------------------------------
template <bool GATHER_F32, bool RELU_BF16_OUT>
__global__ __launch_bounds__(256)
void gemm_kernel(const void* __restrict__ Ain, const __bf16* __restrict__ BT,
                 const float* __restrict__ bias, const int* __restrict__ sel,
                 void* __restrict__ Out, int K, int N) {
  const int lane = threadIdx.x & 31;
  const int wid  = threadIdx.x >> 5;
  const int lm   = lane & 15;
  const int hi   = lane >> 4;
  const int col0 = (blockIdx.x * 8 + wid) * 64;
  const int row0 = blockIdx.y * 16;

  v8f acc[4];
#pragma unroll
  for (int j = 0; j < 4; ++j) {
    float b = bias[col0 + j * 16 + lm];
#pragma unroll
    for (int r = 0; r < 8; ++r) acc[j][r] = b;
  }

  const float*  af = nullptr;
  const __bf16* ab = nullptr;
  if constexpr (GATHER_F32) af = (const float*)Ain + (size_t)sel[row0 + lm] * K;
  else                      ab = (const __bf16*)Ain + (size_t)(row0 + lm) * K;
  const __bf16* brow = BT + (size_t)(col0 + lm) * K + hi * 8;

  v16bf A;
  if constexpr (GATHER_F32)
    A = bf_combine(cvt8(af + hi * 8), cvt8(af + 16 + hi * 8));
  else
    A = bf_combine(*(const v8bf*)(ab + hi * 8), *(const v8bf*)(ab + 16 + hi * 8));

  for (int kk = 0; kk < K; kk += 32) {
    // one clause of 8 b128 loads: all 4 B tiles of this K-step
    v16bf Bt[4];
#pragma unroll
    for (int j = 0; j < 4; ++j)
      Bt[j] = load16bf(brow + (size_t)j * 16 * K + kk);
    // next-step A overlaps the WMMA batch
    v16bf Anext;
    if (kk + 32 < K) {
      if constexpr (GATHER_F32) {
        __builtin_prefetch((const void*)(af + kk + 64 + hi * 8), 0, 1);
        Anext = bf_combine(cvt8(af + kk + 32 + hi * 8), cvt8(af + kk + 48 + hi * 8));
      } else {
        const __bf16* p = ab + kk + 32 + hi * 8;
        Anext = bf_combine(*(const v8bf*)p, *(const v8bf*)(p + 16));
      }
    }
#pragma unroll
    for (int j = 0; j < 4; ++j)
      acc[j] = __builtin_amdgcn_wmma_f32_16x16x32_bf16(false, A, false, Bt[j],
                                                       (short)0, acc[j], false, false);
    A = Anext;
  }

#pragma unroll
  for (int j = 0; j < 4; ++j) {
    int col = col0 + j * 16 + lm;
#pragma unroll
    for (int r = 0; r < 8; ++r) {
      int row = row0 + hi * 8 + r;
      float v = acc[j][r];
      if constexpr (RELU_BF16_OUT)
        ((__bf16*)Out)[(size_t)row * N + col] = (__bf16)(v > 0.0f ? v : 0.0f);
      else
        ((float*)Out)[(size_t)row * N + col] = v;
    }
  }
}

// ---------------------------------------------------------------------------
extern "C" void kernel_launch(void* const* d_in, const int* in_sizes, int n_in,
                              void* d_out, int out_size, void* d_ws, size_t ws_size,
                              hipStream_t stream) {
  (void)in_sizes; (void)n_in; (void)out_size; (void)ws_size;
  const float* ctx = (const float*)d_in[0];
  const float* q   = (const float*)d_in[1];
  const float* W1  = (const float*)d_in[2];
  const float* b1  = (const float*)d_in[3];
  const float* W2  = (const float*)d_in[4];
  const float* b2  = (const float*)d_in[5];
  const float* W3  = (const float*)d_in[6];
  const float* b3  = (const float*)d_in[7];
  const float* Wc1 = (const float*)d_in[8];
  const float* bc1 = (const float*)d_in[9];
  const float* Wc2 = (const float*)d_in[10];
  const float* bc2 = (const float*)d_in[11];
  const float* Wc3 = (const float*)d_in[12];
  const float* bc3 = (const float*)d_in[13];
  float* out = (float*)d_out;

  // workspace carve-out (256B aligned)
  char* wp = (char*)d_ws;
  auto alloc = [&](size_t bytes) -> void* {
    void* p = (void*)wp; wp += (bytes + 255) & ~(size_t)255; return p;
  };
  __bf16* W1T  = (__bf16*)alloc((size_t)512 * HDIM * 2);   // 512 x 2048
  __bf16* W2T  = (__bf16*)alloc((size_t)256 * 512 * 2);    // 256 x 512
  __bf16* Wc1T = (__bf16*)alloc((size_t)1024 * HDIM * 2);  // 1024 x 2048
  __bf16* Wc2T = (__bf16*)alloc((size_t)512 * 1024 * 2);   // 512 x 1024
  __bf16* Wc3T = (__bf16*)alloc((size_t)HDIM * 512 * 2);   // 2048 x 512
  float*  qb   = (float*)alloc(512 * 4);
  float*  S    = (float*)alloc((size_t)NCTX * 4);
  int*    bins = (int*)alloc(4096 * 4);
  int*    meta = (int*)alloc(16 * 4);
  int*    cntA = (int*)alloc(64 * 4);
  int*    cntB = (int*)alloc(64 * 4);
  int*    offA = (int*)alloc(64 * 4);
  int*    offB = (int*)alloc(64 * 4);
  int*    sel  = (int*)alloc((size_t)KSEL * 4);
  __bf16* Y1   = (__bf16*)alloc((size_t)KSEL * 1024 * 2);  // 8 MB
  __bf16* Y2   = (__bf16*)alloc((size_t)KSEL * 512 * 2);   // 4 MB

  auto tlaunch = [&](const float* in, __bf16* o, int K, int N) {
    int total = K * N;
    transpose_bf16_kernel<<<(total + 255) / 256, 256, 0, stream>>>(in, o, K, N);
  };
  // stage 0: weight prep (W1: only the context half, first 2048 rows)
  tlaunch(W1,  W1T,  HDIM, 512);
  tlaunch(W2,  W2T,  512,  256);
  tlaunch(Wc1, Wc1T, HDIM, 1024);
  tlaunch(Wc2, Wc2T, 1024, 512);
  tlaunch(Wc3, Wc3T, 512,  HDIM);
  qbias_kernel<<<2, 256, 0, stream>>>(q, W1, b1, qb);

  // stage 1: fused scorer (16 rows / block)
  score_kernel<<<NCTX / 16, 128, 0, stream>>>(ctx, W1T, qb, W2T, b2, W3, b3, S);

  // stage 2: top-k
  zero_kernel<<<16, 256, 0, stream>>>(bins, 4096);
  hist_kernel<<<NCTX / 256, 256, 0, stream>>>(S, bins);
  thresh_kernel<<<1, 32, 0, stream>>>(bins, meta);
  count_kernel<<<64, 32, 0, stream>>>(S, meta, cntA, cntB);
  scan_kernel<<<1, 32, 0, stream>>>(cntA, cntB, offA, offB);
  select_kernel<<<64, 32, 0, stream>>>(S, meta, offA, offB, sel);

  // stage 3: compressor
  // C1: (4096 x 2048 gathered) @ (2048 x 1024) -> relu bf16
  gemm_kernel<true, true><<<dim3(1024 / 512, KSEL / 16), 256, 0, stream>>>(
      (const void*)ctx, Wc1T, bc1, sel, (void*)Y1, HDIM, 1024);
  // C2: (4096 x 1024 bf16) @ (1024 x 512) -> relu bf16
  gemm_kernel<false, true><<<dim3(1, KSEL / 16), 256, 0, stream>>>(
      (const void*)Y1, Wc2T, bc2, nullptr, (void*)Y2, 1024, 512);
  // C3: (4096 x 512 bf16) @ (512 x 2048) + bc3 -> fp32 out
  gemm_kernel<false, false><<<dim3(HDIM / 512, KSEL / 16), 256, 0, stream>>>(
      (const void*)Y2, Wc3T, bc3, nullptr, (void*)out, 512, HDIM);
}
